// _LatentHamiltonianFunc_15779709845805
// MI455X (gfx1250) — matrix operand order repaired and checked by reading, not compiled, not run on hardware
//
#include <hip/hip_runtime.h>

// ---------------------------------------------------------------------------
// LatentHamiltonianFunc on MI455X (gfx1250), bf16 WMMA mixed precision.
//   B=65536 rows, D=128 (HALF=64), HID=512, A=128.
//   One wave32 per block handles a 16-row slab; all GEMMs use
//   v_wmma_f32_16x16x32_bf16 with f32 accumulation. Elementwise tails use
//   hardware transcendentals (v_exp_f32 / v_log_f32 / v_rcp_f32), branchless.
// ---------------------------------------------------------------------------

#define BATCH 65536
#define DIM   128
#define HALF  64
#define HID   512
#define ADIM  128

typedef __attribute__((ext_vector_type(16))) __bf16 v16bf;
typedef __attribute__((ext_vector_type(8)))  float  v8f;

__device__ __forceinline__ unsigned short f2bf(float x) {
  union { __bf16 h; unsigned short u; } c; c.h = (__bf16)x; return c.u;
}
__device__ __forceinline__ float bf2f(unsigned short u) {
  union { __bf16 h; unsigned short u; } c; c.u = u; return (float)c.h;
}

// Branchless softplus: max(x,0) + log(1 + exp(-|x|)).
// Uses the TRANS pipe (v_exp_f32 + v_log_f32); no exec-mask divergence.
__device__ __forceinline__ float softplus_f(float x) {
  float e = __expf(-fabsf(x));           // (0, 1]
  return fmaxf(x, 0.f) + __logf(1.f + e);
}
// sigmoid via hardware reciprocal (v_rcp_f32).
__device__ __forceinline__ float sigmoid_f(float x) {
  return __builtin_amdgcn_rcpf(1.f + __expf(-x));
}

__device__ __forceinline__ v8f wmma_bf16(v16bf a, v16bf b, v8f c) {
  // D = A(16x32 bf16) x B(32x16 bf16) + C(16x16 f32)
  return __builtin_amdgcn_wmma_f32_16x16x32_bf16(
      /*neg_a=*/false, a, /*neg_b=*/false, b,
      /*c_mod=*/(short)0, c, /*reuse_a=*/false, /*reuse_b=*/false);
}

// B fragment: prepacked fragment-major, 32 lanes x 16 bf16 contiguous.
__device__ __forceinline__ v16bf load_bfrag(const unsigned short* fr, int f, int lane) {
  return ((const v16bf*)fr)[f * 32 + lane];
}

// A fragment (16x32 bf16) from a row-major bf16 LDS tile (16 x HID).
// ISA layout: lane = M + 16*octet, VGPR r<4 -> K 0..7 (+8*octet),
//             VGPR r>=4 -> K 16..23 (+8*octet); 2 K-values per VGPR.
__device__ __forceinline__ v16bf load_a_lds(const unsigned short* buf,
                                            int m, int hi8, int kt) {
  v16bf a;
#pragma unroll
  for (int r = 0; r < 8; ++r) {
    int k = kt * 32 + ((2 * r) & 7) + 8 * hi8 + 16 * (r >> 2);
    unsigned w = *(const unsigned*)(buf + m * HID + k);  // 2 packed bf16
    union { unsigned u; __bf16 h[2]; } cv; cv.u = w;
    a[2 * r] = cv.h[0];
    a[2 * r + 1] = cv.h[1];
  }
  return a;
}

// A fragment from global f32 (z / action_emb), row stride in floats.
__device__ __forceinline__ v16bf load_a_gf32(const float* src, int stride,
                                             int m, int hi8, int kt) {
  v16bf a;
#pragma unroll
  for (int r = 0; r < 8; ++r) {
    int k = kt * 32 + ((2 * r) & 7) + 8 * hi8 + 16 * (r >> 2);
    float2 t = *(const float2*)(src + m * stride + k);
    a[2 * r] = (__bf16)t.x;
    a[2 * r + 1] = (__bf16)t.y;
  }
  return a;
}

// ---------------------------------------------------------------------------
// Weight pack kernel: f32 row-major -> bf16 B-fragments, fragment-major.
// frag f = nt*(K/32)+kt ; element idx = (f*32 + lane)*16 + slot
//   n = nt*16 + (lane&15) ; k = kt*32 + (slot&7) + 8*(lane>>4) + 16*(slot>>3)
// ---------------------------------------------------------------------------
__global__ void pack_bfrags(const float* __restrict__ W,
                            unsigned short* __restrict__ dst,
                            int K, int N, int ld, int transpose) {
  int total = K * N;
  int tilesK = K >> 5;
  for (int idx = blockIdx.x * blockDim.x + threadIdx.x; idx < total;
       idx += gridDim.x * blockDim.x) {
    int slot = idx & 15;
    int lane = (idx >> 4) & 31;
    int f    = idx >> 9;
    int kt = f % tilesK;
    int nt = f / tilesK;
    int k = kt * 32 + (slot & 7) + ((lane >> 4) << 3) + ((slot >> 3) << 4);
    int n = nt * 16 + (lane & 15);
    float v = transpose ? W[n * ld + k] : W[k * ld + n];
    dst[idx] = f2bf(v);
  }
}

// ---------------------------------------------------------------------------
// Main kernel: one wave per block, 16 batch rows per block.
// ---------------------------------------------------------------------------
__global__ __launch_bounds__(32) void hnn_grad_kernel(
    const float* __restrict__ z, const float* __restrict__ act,
    const float* __restrict__ damping,
    const float* __restrict__ b1, const float* __restrict__ b2,
    const float* __restrict__ W3, const float* __restrict__ Gb,
    const unsigned short* __restrict__ w1f,   // W1  : K=128  N=512
    const unsigned short* __restrict__ w2f,   // W2  : K=512  N=512
    const unsigned short* __restrict__ w2tf,  // W2^T: K=512  N=512
    const unsigned short* __restrict__ w1tf,  // W1^T: K=512  N=128
    const unsigned short* __restrict__ gwf,   // Gw  : K=128  N=64
    float* __restrict__ out) {
  __shared__ unsigned short h1buf[16 * HID];  // h1 bf16, later g_a1 bf16
  __shared__ unsigned short gbuf[16 * HID];   // g_a2 bf16

  const int lane = threadIdx.x;        // 0..31
  const int m    = lane & 15;          // A-matrix row owned by this lane
  const int hi8  = lane >> 4;          // K-octet select (A) / row+8 (D)
  const int ncol = lane & 15;          // D-layout column
  const int row0 = blockIdx.x * 16;

  // ---- GEMM1: a1 = z @ W1 + b1 ; h1 = softplus(a1) -> LDS bf16 ----------
  v16bf za[4];
#pragma unroll
  for (int kt = 0; kt < 4; ++kt)
    za[kt] = load_a_gf32(z + row0 * DIM, DIM, m, hi8, kt);

  for (int nt = 0; nt < 32; ++nt) {
    __builtin_prefetch(w1f + ((nt + 1) & 31) * 4 * 512, 0, 0);
    v8f acc = {};
#pragma unroll
    for (int kt = 0; kt < 4; ++kt)
      acc = wmma_bf16(za[kt], load_bfrag(w1f, nt * 4 + kt, lane), acc);
    int col = nt * 16 + ncol;
    float bias = b1[col];
#pragma unroll
    for (int r = 0; r < 8; ++r) {
      int M = r + 8 * hi8;
      h1buf[M * HID + col] = f2bf(softplus_f(acc[r] + bias));
    }
  }
  __syncthreads();

  // ---- GEMM2: a2 = h1 @ W2 + b2 ; g_a2 = sigmoid(a2)*W3 -> LDS bf16 -----
  v16bf ha[16];
#pragma unroll
  for (int kt = 0; kt < 16; ++kt) ha[kt] = load_a_lds(h1buf, m, hi8, kt);

  for (int nt = 0; nt < 32; ++nt) {
    __builtin_prefetch(w2f + ((nt + 1) & 31) * 16 * 512, 0, 0);
    v8f acc = {};
#pragma unroll
    for (int kt = 0; kt < 16; ++kt)
      acc = wmma_bf16(ha[kt], load_bfrag(w2f, nt * 16 + kt, lane), acc);
    int col = nt * 16 + ncol;
    float bias = b2[col];
    float w3v  = W3[col];
#pragma unroll
    for (int r = 0; r < 8; ++r) {
      int M = r + 8 * hi8;
      gbuf[M * HID + col] = f2bf(sigmoid_f(acc[r] + bias) * w3v);
    }
  }
  __syncthreads();

  // ---- GEMM3: g_h1 = g_a2 @ W2^T ; g_a1 = s1*g_h1, s1 = 1-exp(-h1) ------
  // g_a1 overwrites h1buf in place (tile n only needs h1 columns of tile n).
  v16bf ga[16];
#pragma unroll
  for (int kt = 0; kt < 16; ++kt) ga[kt] = load_a_lds(gbuf, m, hi8, kt);

  for (int nt = 0; nt < 32; ++nt) {
    __builtin_prefetch(w2tf + ((nt + 1) & 31) * 16 * 512, 0, 0);
    v8f acc = {};
#pragma unroll
    for (int kt = 0; kt < 16; ++kt)
      acc = wmma_bf16(ga[kt], load_bfrag(w2tf, nt * 16 + kt, lane), acc);
    int col = nt * 16 + ncol;
#pragma unroll
    for (int r = 0; r < 8; ++r) {
      int M = r + 8 * hi8;
      float h1v = bf2f(h1buf[M * HID + col]);
      float s1  = 1.f - __expf(-h1v);        // sigmoid(a1) from softplus(a1)
      h1buf[M * HID + col] = f2bf(s1 * acc[r]);
    }
  }
  __syncthreads();

  // ---- GEMM4: dH = g_a1 @ W1^T  (16 x 128, kept in registers) -----------
  v16bf g1a[16];
#pragma unroll
  for (int kt = 0; kt < 16; ++kt) g1a[kt] = load_a_lds(h1buf, m, hi8, kt);

  v8f dH[8];
#pragma unroll
  for (int mt = 0; mt < 8; ++mt) {
    v8f acc = {};
#pragma unroll
    for (int kt = 0; kt < 16; ++kt)
      acc = wmma_bf16(g1a[kt], load_bfrag(w1tf, mt * 16 + kt, lane), acc);
    dH[mt] = acc;
  }

  // ---- GEMM5: G_u = action_emb @ Gw + Gb  (16 x 64) ---------------------
  v16bf aa[4];
#pragma unroll
  for (int kt = 0; kt < 4; ++kt)
    aa[kt] = load_a_gf32(act + row0 * ADIM, ADIM, m, hi8, kt);

  v8f gu[4];
#pragma unroll
  for (int nt = 0; nt < 4; ++nt) {
    v8f acc = {};
#pragma unroll
    for (int kt = 0; kt < 4; ++kt)
      acc = wmma_bf16(aa[kt], load_bfrag(gwf, nt * 4 + kt, lane), acc);
    float gb = Gb[nt * 16 + ncol];
#pragma unroll
    for (int r = 0; r < 8; ++r) acc[r] += gb;
    gu[nt] = acc;
  }

  // ---- Combine: out = [ dH_dp , -dH_dq - damping*dH_dp + G_u ] ----------
  float damp = damping[0];
#pragma unroll
  for (int nt = 0; nt < 4; ++nt) {
    int col = nt * 16 + ncol;
#pragma unroll
    for (int r = 0; r < 8; ++r) {
      int row = row0 + r + 8 * hi8;
      float dq = dH[nt + 4][r];                         // dH_dp column
      float dp = -dH[nt][r] - damp * dq + gu[nt][r];
      out[row * DIM + col] = dq;
      out[row * DIM + HALF + col] = dp;
    }
  }
}

// ---------------------------------------------------------------------------
extern "C" void kernel_launch(void* const* d_in, const int* in_sizes, int n_in,
                              void* d_out, int out_size, void* d_ws, size_t ws_size,
                              hipStream_t stream) {
  (void)in_sizes; (void)n_in; (void)out_size; (void)ws_size;
  const float* z       = (const float*)d_in[0];
  const float* act     = (const float*)d_in[1];
  const float* damping = (const float*)d_in[2];
  const float* W1      = (const float*)d_in[3];
  const float* b1      = (const float*)d_in[4];
  const float* W2      = (const float*)d_in[5];
  const float* b2      = (const float*)d_in[6];
  const float* W3      = (const float*)d_in[7];
  // d_in[8] = b3: constant shift of H, no effect on grad(H) -> unused.
  const float* Gw      = (const float*)d_in[9];
  const float* Gb      = (const float*)d_in[10];
  float* out = (float*)d_out;

  // Workspace layout (bf16 fragment-major weights, ~1.27 MB total).
  char* ws = (char*)d_ws;
  unsigned short* w1f  = (unsigned short*)(ws);                 // 128 KB
  unsigned short* w2f  = (unsigned short*)(ws + (128u << 10));  // 512 KB
  unsigned short* w2tf = (unsigned short*)(ws + (640u << 10));  // 512 KB
  unsigned short* w1tf = (unsigned short*)(ws + (1152u << 10)); // 128 KB
  unsigned short* gwf  = (unsigned short*)(ws + (1280u << 10)); //  16 KB

  pack_bfrags<<<256, 256, 0, stream>>>(W1, w1f,  DIM, HID, HID, 0);
  pack_bfrags<<<1024, 256, 0, stream>>>(W2, w2f,  HID, HID, HID, 0);
  pack_bfrags<<<1024, 256, 0, stream>>>(W2, w2tf, HID, HID, HID, 1);
  pack_bfrags<<<256, 256, 0, stream>>>(W1, w1tf, HID, DIM, HID, 1);
  pack_bfrags<<<32,  256, 0, stream>>>(Gw, gwf,  ADIM, HALF, HALF, 0);

  hnn_grad_kernel<<<BATCH / 16, 32, 0, stream>>>(
      z, act, damping, b1, b2, W3, Gb, w1f, w2f, w2tf, w1tf, gwf, out);
}
